// FNO1D_4741643895631
// MI455X (gfx1250) — compile-verified, matrix operand order checked by reading, not run on hardware
//
#include <hip/hip_runtime.h>

#define B_ 16
#define S_ 8192
#define W_ 128
#define MODES_ 32
#define NLAYER_ 4
#define NEG_ 0.01f
#define KSPLIT_ 8

typedef __bf16 bf16_t;
typedef __attribute__((ext_vector_type(16))) __bf16 v16bf;
typedef __attribute__((ext_vector_type(8)))  float  v8f;
typedef __attribute__((ext_vector_type(4)))  unsigned int u32x4;
typedef __attribute__((ext_vector_type(4)))  int i32x4;
typedef __attribute__((ext_vector_type(8)))  int i32x8;

#if defined(__has_builtin)
#if __has_builtin(__builtin_amdgcn_tensor_load_to_lds) && __has_builtin(__builtin_amdgcn_s_wait_tensorcnt)
#define HAVE_TDM 1
#endif
#endif

#define WMMA(a, b, c) \
  __builtin_amdgcn_wmma_f32_16x16x32_bf16(false, (a), false, (b), (short)0, (c), false, false)

__device__ __forceinline__ float lrelu(float v) { return v < 0.0f ? NEG_ * v : v; }

__device__ __forceinline__ v16bf frag_from(uint4 lo, uint4 hi) {
  union { uint4 u[2]; v16bf v; } u;
  u.u[0] = lo; u.u[1] = hi;
  return u.v;
}

// ---- WMMA fragment loaders (CDNA5 ISA 7.12.2, wave32) ----
// A (MxK) and B (KxN stored as N rows of contiguous K): lane&15 selects row,
// lane>=16 shifts K base by 8; elems 0..7 = K kb..kb+7, 8..15 = kb+16..kb+23.
// ld must be a multiple of 8 elements -> two b128 loads per lane.
__device__ __forceinline__ v16bf load_frag_rm(const bf16_t* src, int ld, int row0, int k0) {
  int lane = threadIdx.x & 31;
  const bf16_t* p = src + (size_t)(row0 + (lane & 15)) * ld + k0 + ((lane >> 4) << 3);
  uint4 lo = *(const uint4*)p;
  uint4 hi = *(const uint4*)(p + 16);
  return frag_from(lo, hi);
}

__device__ __forceinline__ v16bf load_frag_rm_f32(const float* src, int ld, int row0, int k0) {
  int lane = threadIdx.x & 31;
  const float* p = src + (size_t)(row0 + (lane & 15)) * ld + k0 + ((lane >> 4) << 3);
  float4 x0 = *(const float4*)p;
  float4 x1 = *(const float4*)(p + 4);
  float4 x2 = *(const float4*)(p + 16);
  float4 x3 = *(const float4*)(p + 20);
  v16bf v;
  v[0]  = (bf16_t)x0.x; v[1]  = (bf16_t)x0.y; v[2]  = (bf16_t)x0.z; v[3]  = (bf16_t)x0.w;
  v[4]  = (bf16_t)x1.x; v[5]  = (bf16_t)x1.y; v[6]  = (bf16_t)x1.z; v[7]  = (bf16_t)x1.w;
  v[8]  = (bf16_t)x2.x; v[9]  = (bf16_t)x2.y; v[10] = (bf16_t)x2.z; v[11] = (bf16_t)x2.w;
  v[12] = (bf16_t)x3.x; v[13] = (bf16_t)x3.y; v[14] = (bf16_t)x3.z; v[15] = (bf16_t)x3.w;
  return v;
}

// D (16x16 f32): VGPR r: lanes<16 -> M=r, N=lane ; lanes>=16 -> M=r+8, N=lane-16.
__device__ __forceinline__ void store_d(float* dst, int ld, int row0, int col0, v8f acc) {
  int lane = threadIdx.x & 31;
  int n = col0 + (lane & 15);
  int mb = (lane >> 4) << 3;
#pragma unroll
  for (int r = 0; r < 8; ++r) dst[(size_t)(row0 + mb + r) * ld + n] = acc[r];
}

// ---- basis tables ----
__global__ void k_basis(bf16_t* __restrict__ bFt, bf16_t* __restrict__ bIT) {
  int s = blockIdx.x * blockDim.x + threadIdx.x;
  if (s >= S_) return;
  const float TWO_PI = 6.283185307179586f;
  for (int k = 0; k < MODES_; ++k) {
    int ph = (k * s) & (S_ - 1);
    float ang = TWO_PI * (float)ph / (float)S_;
    float c = cosf(ang), sn = sinf(ang);
    bFt[(size_t)k * S_ + s]        = (bf16_t)c;
    bFt[(size_t)(32 + k) * S_ + s] = (bf16_t)(-sn);
    float sc = (k == 0) ? (1.0f / S_) : (2.0f / S_);
    bIT[(size_t)s * 64 + k]      = (bf16_t)(sc * c);
    bIT[(size_t)s * 64 + 32 + k] = (bf16_t)((k == 0) ? 0.0f : (-sc * sn));
  }
}

// ---- transpose spectral weights to K-contiguous (layer, k, o, i) bf16 ----
__global__ void k_wT(const float* __restrict__ wr, const float* __restrict__ wi,
                     bf16_t* __restrict__ wTr, bf16_t* __restrict__ wTi) {
  int idx = blockIdx.x * 256 + threadIdx.x;      // NLAYER*32*128*128
  int k = idx & 31, o = (idx >> 5) & 127, i = (idx >> 12) & 127, l = idx >> 19;
  size_t srcoff = (size_t)l * W_ * W_ * MODES_ + (size_t)(i * 128 + o) * 32 + k;
  size_t dstoff = (size_t)l * MODES_ * W_ * W_ + ((size_t)k * 128 + o) * 128 + i;
  wTr[dstoff] = (bf16_t)wr[srcoff];
  wTi[dstoff] = (bf16_t)wi[srcoff];
}

// ---- lift ----
__global__ void k_lift(const float* __restrict__ x, const float* __restrict__ Pw,
                       const float* __restrict__ Pb, bf16_t* __restrict__ h) {
  long idx = (long)blockIdx.x * blockDim.x + threadIdx.x;
  int s = (int)(idx & (S_ - 1));
  int row = (int)(idx >> 13);
  int w = row & (W_ - 1);
  int b = row >> 7;
  h[idx] = (bf16_t)(Pw[w] * x[b * S_ + s] + Pb[w]);
}

// ---- forward DFT, split-K: part[kz](64x2048) = bFt(64xKs) * h^T(Ksx2048) ----
__global__ void __launch_bounds__(256) k_fwd(const bf16_t* __restrict__ h,
                                             const bf16_t* __restrict__ bFt,
                                             float* __restrict__ part) {
  int wave = threadIdx.x >> 5;
  int mrow0 = (wave & 3) * 16;
  int ncol0 = blockIdx.x * 128 + (wave >> 2) * 64;
  int kbeg = blockIdx.y << 10;                       // 1024-wide K slice
  v8f a0 = {}, a1 = {}, a2 = {}, a3 = {};
  for (int k0 = kbeg; k0 < kbeg + 1024; k0 += 32) {
    __builtin_prefetch(h + (size_t)ncol0 * S_ + k0 + 256, 0, 1);
    v16bf a  = load_frag_rm(bFt, S_, mrow0, k0);
    v16bf b0 = load_frag_rm(h, S_, ncol0 + 0,  k0);
    v16bf b1 = load_frag_rm(h, S_, ncol0 + 16, k0);
    v16bf b2 = load_frag_rm(h, S_, ncol0 + 32, k0);
    v16bf b3 = load_frag_rm(h, S_, ncol0 + 48, k0);
    a0 = WMMA(a, b0, a0);
    a1 = WMMA(a, b1, a1);
    a2 = WMMA(a, b2, a2);
    a3 = WMMA(a, b3, a3);
  }
  float* dst = part + (size_t)blockIdx.y * 64 * 2048;
  store_d(dst, 2048, mrow0, ncol0 + 0,  a0);
  store_d(dst, 2048, mrow0, ncol0 + 16, a1);
  store_d(dst, 2048, mrow0, ncol0 + 32, a2);
  store_d(dst, 2048, mrow0, ncol0 + 48, a3);
}

// deterministic split-K reduce
__global__ void k_red(const float* __restrict__ part, float* __restrict__ ftcT) {
  int idx = blockIdx.x * 256 + threadIdx.x;          // 64*2048
  float s = 0.0f;
#pragma unroll
  for (int z = 0; z < KSPLIT_; ++z) s += part[(size_t)z * 64 * 2048 + idx];
  ftcT[idx] = s;
}

// ---- per-mode complex mix on WMMA: D(16x128) = F(16x128) * W(128x128) per mode ----
// Dr = Fr*Wr + (-Fi)*Wi ; Di = Fr*Wi + Fi*Wr  (bf16 WMMA has no A-negation -> negate Fi)
__global__ void __launch_bounds__(64) k_mix(const float* __restrict__ ftcT,
                                            const bf16_t* __restrict__ wTr,
                                            const bf16_t* __restrict__ wTi,
                                            bf16_t* __restrict__ gm) {
  int k = blockIdx.x;                                // mode
  int n0 = (threadIdx.x >> 5) * 64;                  // o-column half
  const float* fr = ftcT + (size_t)k * 2048;
  const float* fi = ftcT + (size_t)(32 + k) * 2048;
  const bf16_t* br = wTr + (size_t)k * 128 * 128;
  const bf16_t* bi = wTi + (size_t)k * 128 * 128;
  v8f dr0 = {}, dr1 = {}, dr2 = {}, dr3 = {};
  v8f di0 = {}, di1 = {}, di2 = {}, di3 = {};
#pragma unroll
  for (int k0 = 0; k0 < 128; k0 += 32) {
    v16bf Fr = load_frag_rm_f32(fr, 128, 0, k0);     // rows = batches
    v16bf Fi = load_frag_rm_f32(fi, 128, 0, k0);
    v16bf Fni;
#pragma unroll
    for (int e = 0; e < 16; ++e) Fni[e] = (bf16_t)(-(float)Fi[e]);
    v16bf Br0 = load_frag_rm(br, 128, n0 + 0,  k0);
    v16bf Br1 = load_frag_rm(br, 128, n0 + 16, k0);
    v16bf Br2 = load_frag_rm(br, 128, n0 + 32, k0);
    v16bf Br3 = load_frag_rm(br, 128, n0 + 48, k0);
    v16bf Bi0 = load_frag_rm(bi, 128, n0 + 0,  k0);
    v16bf Bi1 = load_frag_rm(bi, 128, n0 + 16, k0);
    v16bf Bi2 = load_frag_rm(bi, 128, n0 + 32, k0);
    v16bf Bi3 = load_frag_rm(bi, 128, n0 + 48, k0);
    dr0 = WMMA(Fr, Br0, dr0); dr0 = WMMA(Fni, Bi0, dr0);
    dr1 = WMMA(Fr, Br1, dr1); dr1 = WMMA(Fni, Bi1, dr1);
    dr2 = WMMA(Fr, Br2, dr2); dr2 = WMMA(Fni, Bi2, dr2);
    dr3 = WMMA(Fr, Br3, dr3); dr3 = WMMA(Fni, Bi3, dr3);
    di0 = WMMA(Fr, Bi0, di0); di0 = WMMA(Fi, Br0, di0);
    di1 = WMMA(Fr, Bi1, di1); di1 = WMMA(Fi, Br1, di1);
    di2 = WMMA(Fr, Bi2, di2); di2 = WMMA(Fi, Br2, di2);
    di3 = WMMA(Fr, Bi3, di3); di3 = WMMA(Fi, Br3, di3);
  }
  int lane = threadIdx.x & 31;
  int on = lane & 15;
  int mb = (lane >> 4) << 3;
#pragma unroll
  for (int r = 0; r < 8; ++r) {
    int b = mb + r;                                  // batch (D row)
    bf16_t* g = gm + (size_t)b * 128 * 64;
    g[(n0 + 0  + on) * 64 + k] = (bf16_t)dr0[r];
    g[(n0 + 16 + on) * 64 + k] = (bf16_t)dr1[r];
    g[(n0 + 32 + on) * 64 + k] = (bf16_t)dr2[r];
    g[(n0 + 48 + on) * 64 + k] = (bf16_t)dr3[r];
    g[(n0 + 0  + on) * 64 + 32 + k] = (bf16_t)di0[r];
    g[(n0 + 16 + on) * 64 + 32 + k] = (bf16_t)di1[r];
    g[(n0 + 32 + on) * 64 + 32 + k] = (bf16_t)di2[r];
    g[(n0 + 48 + on) * 64 + 32 + k] = (bf16_t)di3[r];
  }
}

// ---- fused: h' = lrelu( irfft_modes(gm) + conv3(h) + cb ) per (batch, 64-col tile) ----
__global__ void __launch_bounds__(256) k_fused(
    const bf16_t* __restrict__ h, const bf16_t* __restrict__ gm,
    const bf16_t* __restrict__ bIT, const float* __restrict__ cw,
    const float* __restrict__ cb, bf16_t* __restrict__ hn) {
  __shared__ bf16_t s_imT[64 * 392];   // im2col^T: [c][i*3+t], ld 392 (aligned + bank-spread)
  __shared__ bf16_t s_biT[64 * 72];    // irFFT basis tile: [c][k], ld 72 = TDM pad 128B+16B
  int s0 = blockIdx.x * 64;
  int b  = blockIdx.y;
  int tid = threadIdx.x;

#ifdef HAVE_TDM
  // Tensor Data Mover: 64x64 2-byte tile of bIT starting at row s0, LDS row
  // pitch 144B via pad_interval=32 dwords (128B), pad_amount=4 dwords (16B).
  if (tid == 0) {
    unsigned long long ga = (unsigned long long)(const void*)(bIT + (size_t)s0 * 64);
    unsigned int lds_a = (unsigned int)(unsigned long long)(void*)s_biT;
    u32x4 g0 = {};
    g0[0] = 1u;                                               // count=1 user descriptor
    g0[1] = lds_a;                                            // lds_addr
    g0[2] = (unsigned int)ga;                                 // global_addr[31:0]
    g0[3] = (unsigned int)((ga >> 32) & 0x01FFFFFFull) | (2u << 30);  // addr[56:32]|type=2
    i32x8 g1 = {};
    g1[0] = (1 << 16) | (1 << 20) | (4 << 22) | (3 << 25);    // dsz=2B, pad: 32dw + 4dw
    g1[1] = (64 << 16);                                       // tensor_dim0 = 64
    g1[2] = (64 & 0xFFFF) << 16;                              // tensor_dim1 (lo16) = 64
    g1[3] = (64 << 16);                                       // tile_dim0 = 64
    g1[4] = 64;                                               // tile_dim1 = 64
    g1[5] = 64;                                               // tensor_dim0_stride = 64
    i32x4 g2 = {}, g3 = {};
#if defined(__clang_major__) && __clang_major__ >= 23
    i32x8 g4 = {};
    __builtin_amdgcn_tensor_load_to_lds(g0, g1, g2, g3, g4, 0);
#else
    __builtin_amdgcn_tensor_load_to_lds(g0, g1, g2, g3, 0);
#endif
  }
#else
  for (int idx = tid; idx < 64 * 64; idx += 256) {
    int k = idx & 63, c = idx >> 6;
    s_biT[c * 72 + k] = bIT[(size_t)(s0 + c) * 64 + k];
  }
#endif

  for (int idx = tid; idx < 64 * 128; idx += 256) {
    int c = idx & 63, i = idx >> 6;
    const bf16_t* hrow = h + (size_t)(b * 128 + i) * S_;
    bf16_t* dst = &s_imT[c * 392 + 3 * i];
    int sbase = s0 + c - 1;
#pragma unroll
    for (int t = 0; t < 3; ++t) {
      int sg = sbase + t;
      dst[t] = (sg >= 0 && sg < S_) ? hrow[sg] : (bf16_t)0.0f;
    }
  }
#ifdef HAVE_TDM
  __builtin_amdgcn_s_wait_tensorcnt(0);   // waves with TENSORcnt==0 pass immediately
#endif
  __syncthreads();

  int wave = tid >> 5;
  int orow0 = wave * 16;
  const bf16_t* gmb = gm + (size_t)b * 128 * 64;
  v8f a0 = {}, a1 = {}, a2 = {}, a3 = {};

#pragma unroll
  for (int k0 = 0; k0 < 64; k0 += 32) {              // spectral: K = 64
    v16bf a  = load_frag_rm(gmb, 64, orow0, k0);
    v16bf b0 = load_frag_rm(s_biT, 72, 0,  k0);
    v16bf b1 = load_frag_rm(s_biT, 72, 16, k0);
    v16bf b2 = load_frag_rm(s_biT, 72, 32, k0);
    v16bf b3 = load_frag_rm(s_biT, 72, 48, k0);
    a0 = WMMA(a, b0, a0);
    a1 = WMMA(a, b1, a1);
    a2 = WMMA(a, b2, a2);
    a3 = WMMA(a, b3, a3);
  }
  for (int k0 = 0; k0 < 384; k0 += 32) {             // conv: K = 384
    v16bf a  = load_frag_rm_f32(cw, 384, orow0, k0);
    v16bf b0 = load_frag_rm(s_imT, 392, 0,  k0);
    v16bf b1 = load_frag_rm(s_imT, 392, 16, k0);
    v16bf b2 = load_frag_rm(s_imT, 392, 32, k0);
    v16bf b3 = load_frag_rm(s_imT, 392, 48, k0);
    a0 = WMMA(a, b0, a0);
    a1 = WMMA(a, b1, a1);
    a2 = WMMA(a, b2, a2);
    a3 = WMMA(a, b3, a3);
  }

  int lane = tid & 31;
  int n = lane & 15;
  int mb = (lane >> 4) << 3;
#pragma unroll
  for (int r = 0; r < 8; ++r) {
    int orow = orow0 + mb + r;
    float bias = cb[orow];
    bf16_t* dst = hn + (size_t)(b * 128 + orow) * S_ + s0 + n;
    dst[0]  = (bf16_t)lrelu(a0[r] + bias);
    dst[16] = (bf16_t)lrelu(a1[r] + bias);
    dst[32] = (bf16_t)lrelu(a2[r] + bias);
    dst[48] = (bf16_t)lrelu(a3[r] + bias);
  }
}

// ---- projection: out[b,s] = Q2 . lrelu(Q1 h[:,s] + b1) + b2 (deterministic reduce) ----
__global__ void __launch_bounds__(256) k_proj(
    const bf16_t* __restrict__ h, const float* __restrict__ Q1w,
    const float* __restrict__ Q1b, const float* __restrict__ Q2w,
    const float* __restrict__ Q2b, float* __restrict__ out) {
  __shared__ bf16_t s_hT[64 * 136];    // [c][w], ld 136 (aligned + bank-spread)
  __shared__ float s_part[16][64];     // per-(wave, lane-half) partials, fixed-order sum
  int s0 = blockIdx.x * 64;
  int b  = blockIdx.y;
  int tid = threadIdx.x;

  for (int idx = tid; idx < 64 * 128; idx += 256) {
    int c = idx & 63, w = idx >> 6;
    s_hT[c * 136 + w] = h[(size_t)(b * 128 + w) * S_ + s0 + c];
  }
  __syncthreads();

  int wave = tid >> 5;
  int j0 = wave * 16;
  v8f a0 = {}, a1 = {}, a2 = {}, a3 = {};
#pragma unroll
  for (int k0 = 0; k0 < 128; k0 += 32) {
    v16bf a  = load_frag_rm_f32(Q1w, 128, j0, k0);
    v16bf b0 = load_frag_rm(s_hT, 136, 0,  k0);
    v16bf b1 = load_frag_rm(s_hT, 136, 16, k0);
    v16bf b2 = load_frag_rm(s_hT, 136, 32, k0);
    v16bf b3 = load_frag_rm(s_hT, 136, 48, k0);
    a0 = WMMA(a, b0, a0);
    a1 = WMMA(a, b1, a1);
    a2 = WMMA(a, b2, a2);
    a3 = WMMA(a, b3, a3);
  }

  int lane = tid & 31;
  int n = lane & 15;
  int hb = lane >> 4;
  int mb = hb << 3;
  float p0 = 0, p1 = 0, p2 = 0, p3 = 0;
#pragma unroll
  for (int r = 0; r < 8; ++r) {
    int j = j0 + mb + r;
    float bj = Q1b[j], q2 = Q2w[j];
    p0 += lrelu(a0[r] + bj) * q2;
    p1 += lrelu(a1[r] + bj) * q2;
    p2 += lrelu(a2[r] + bj) * q2;
    p3 += lrelu(a3[r] + bj) * q2;
  }
  int slot = wave * 2 + hb;            // unique (wave, half) writer per column
  s_part[slot][n +  0] = p0;
  s_part[slot][n + 16] = p1;
  s_part[slot][n + 32] = p2;
  s_part[slot][n + 48] = p3;
  __syncthreads();
  if (tid < 64) {
    float s = 0.0f;
#pragma unroll
    for (int q = 0; q < 16; ++q) s += s_part[q][tid];
    out[(size_t)b * S_ + s0 + tid] = s + Q2b[0];
  }
}

extern "C" void kernel_launch(void* const* d_in, const int* in_sizes, int n_in,
                              void* d_out, int out_size, void* d_ws, size_t ws_size,
                              hipStream_t stream) {
  (void)in_sizes; (void)n_in; (void)out_size; (void)ws_size;
  const float* x   = (const float*)d_in[0];
  const float* Pw  = (const float*)d_in[1];
  const float* Pb  = (const float*)d_in[2];
  const float* swr = (const float*)d_in[3];
  const float* swi = (const float*)d_in[4];
  const float* cw  = (const float*)d_in[5];
  const float* cb  = (const float*)d_in[6];
  const float* Q1w = (const float*)d_in[7];
  const float* Q1b = (const float*)d_in[8];
  const float* Q2w = (const float*)d_in[9];
  const float* Q2b = (const float*)d_in[10];
  float* out = (float*)d_out;

  char* ws = (char*)d_ws;
  size_t off = 0;
  auto alloc = [&](size_t bytes) -> char* {
    char* p = ws + off;
    off += (bytes + 255) & ~(size_t)255;
    return p;
  };
  bf16_t* hA   = (bf16_t*)alloc((size_t)B_ * W_ * S_ * 2);            // 32 MB
  bf16_t* hB   = (bf16_t*)alloc((size_t)B_ * W_ * S_ * 2);            // 32 MB
  bf16_t* bFt  = (bf16_t*)alloc((size_t)64 * S_ * 2);                 //  1 MB
  bf16_t* bIT  = (bf16_t*)alloc((size_t)S_ * 64 * 2);                 //  1 MB
  float*  ftcT = (float*)alloc((size_t)64 * B_ * W_ * 4);             // .5 MB
  bf16_t* gm   = (bf16_t*)alloc((size_t)B_ * W_ * 64 * 2);            // .25 MB
  float*  part = (float*)alloc((size_t)KSPLIT_ * 64 * B_ * W_ * 4);   //  4 MB
  bf16_t* wTr  = (bf16_t*)alloc((size_t)NLAYER_ * MODES_ * W_ * W_ * 2);  // 4 MB
  bf16_t* wTi  = (bf16_t*)alloc((size_t)NLAYER_ * MODES_ * W_ * W_ * 2);  // 4 MB

  k_basis<<<S_ / 256, 256, 0, stream>>>(bFt, bIT);
  k_wT<<<(NLAYER_ * MODES_ * W_ * W_) / 256, 256, 0, stream>>>(swr, swi, wTr, wTi);
  k_lift<<<(B_ * W_ * S_) / 256, 256, 0, stream>>>(x, Pw, Pb, hA);

  bf16_t* cur = hA;
  bf16_t* nxt = hB;
  for (int L = 0; L < NLAYER_; ++L) {
    k_fwd<<<dim3((B_ * W_) / 128, KSPLIT_), 256, 0, stream>>>(cur, bFt, part);
    k_red<<<(64 * B_ * W_) / 256, 256, 0, stream>>>(part, ftcT);
    k_mix<<<MODES_, 64, 0, stream>>>(
        ftcT, wTr + (size_t)L * MODES_ * W_ * W_, wTi + (size_t)L * MODES_ * W_ * W_, gm);
    k_fused<<<dim3(S_ / 64, B_), 256, 0, stream>>>(
        cur, gm, bIT, cw + (size_t)L * W_ * W_ * 3, cb + (size_t)L * W_, nxt);
    bf16_t* t = cur; cur = nxt; nxt = t;
  }

  k_proj<<<dim3(S_ / 64, B_), 256, 0, stream>>>(cur, Q1w, Q1b, Q2w, Q2b, out);
}
